// MultiheadAttention_39676907886922
// MI455X (gfx1250) — compile-verified, matrix operand order checked by reading, not run on hardware
//
#include <hip/hip_runtime.h>
#include <hip/hip_bf16.h>

// ---------------------------------------------------------------------------
// MI455X (gfx1250) fused multi-head self-attention.
// v_wmma_f32_16x16x32_bf16 everywhere; bf16 workspaces; attention K/V tiles
// double-buffered in LDS via the Tensor Data Mover (fallback: async-to-LDS).
// T=2048, B=2, E=1024, H=16, hd=64.  Wave32, 8 waves / 256-thread block.
// ---------------------------------------------------------------------------

typedef __bf16 bf16_t;
typedef bf16_t bf16x16 __attribute__((ext_vector_type(16)));
typedef bf16_t bf16x8  __attribute__((ext_vector_type(8)));
typedef float  f32x8   __attribute__((ext_vector_type(8)));
typedef float  f32x4   __attribute__((ext_vector_type(4)));
typedef int    i32x4   __attribute__((ext_vector_type(4)));
typedef int    i32x8   __attribute__((ext_vector_type(8)));
typedef unsigned u32x4 __attribute__((ext_vector_type(4)));

#define T_LEN 2048
#define EMB 1024
#define NHEAD 16
#define HD 64
#define BH 32
#define NEG_INF (-__builtin_inff())

#define AS1 __attribute__((address_space(1)))
#define AS3 __attribute__((address_space(3)))

#if __has_builtin(__builtin_amdgcn_tensor_load_to_lds)
#define HAVE_TDM 1
#else
#define HAVE_TDM 0
#endif
#if __has_builtin(__builtin_amdgcn_global_load_async_to_lds_b128)
#define HAVE_ASYNC_LDS 1
#else
#define HAVE_ASYNC_LDS 0
#endif

// LDS byte offset of a generic shared-memory pointer.
__device__ __forceinline__ unsigned lds_offset(void* p) {
  return (unsigned)(unsigned long long)(AS3 void*)p;
}

#if HAVE_TDM
// Issue a TDM 2-D tile load: rows x rowBytes (1-byte elements), global row
// stride strideBytes, packed contiguously into LDS at lds_off.
// D# groups assembled per CDNA5 ISA 8.3/8.4; 6-arg builtin (clang-23 flavor).
__device__ __forceinline__ void tdm_load_2d(unsigned lds_off,
                                            unsigned long long gaddr,
                                            unsigned rowBytes, unsigned rows,
                                            unsigned strideBytes) {
  u32x4 g0;
  g0[0] = 1u;                                          // count=1 (valid user D#)
  g0[1] = lds_off;                                     // lds_addr
  g0[2] = (unsigned)(gaddr & 0xffffffffu);             // global_addr[31:0]
  g0[3] = (unsigned)((gaddr >> 32) & 0x1ffffffu)       // global_addr[56:32]
          | (2u << 30);                                // type = 2 ("image")
  i32x8 g1;
  g1[0] = 0;                                           // mask=0, data_size=1B
  g1[1] = (int)((rowBytes & 0xffffu) << 16);           // tensor_dim0[15:0]
  g1[2] = (int)((rowBytes >> 16) & 0xffffu)            // tensor_dim0[31:16]
          | (int)((rows & 0xffffu) << 16);             // tensor_dim1[15:0]
  g1[3] = (int)((rows >> 16) & 0xffffu)                // tensor_dim1[31:16]
          | (int)((rowBytes & 0xffffu) << 16);         // tile_dim0
  g1[4] = (int)(rows & 0xffffu);                       // tile_dim1 (tile_dim2=0)
  g1[5] = (int)strideBytes;                            // tensor_dim0_stride lo
  g1[6] = 0;                                           // stride hi / dim1_stride
  g1[7] = 0;
  const i32x4 z4 = {0, 0, 0, 0};
  const i32x8 z8 = {0, 0, 0, 0, 0, 0, 0, 0};
  __builtin_amdgcn_tensor_load_to_lds(g0, g1, z4, z4, z8, 0);
}
#endif

// 16-byte global -> LDS copy (async fallback path).
__device__ __forceinline__ void copy16_to_lds(const bf16_t* g, bf16_t* l) {
#if HAVE_ASYNC_LDS
  __builtin_amdgcn_global_load_async_to_lds_b128(
      (AS1 i32x4*)(AS1 void*)(void*)g,
      (AS3 i32x4*)(AS3 void*)(void*)l, 0, 0);
#else
  *(bf16x8*)l = *(const bf16x8*)g;
#endif
}

__device__ __forceinline__ void wait_async_copies() {
#if HAVE_ASYNC_LDS
#if __has_builtin(__builtin_amdgcn_s_wait_asynccnt)
  __builtin_amdgcn_s_wait_asynccnt(0);
#else
  asm volatile("s_wait_asynccnt 0x0" ::: "memory");
#endif
#endif
}

// ---- WMMA wrapper ----------------------------------------------------------
__device__ __forceinline__ f32x8 wmma_bf16(bf16x16 a, bf16x16 b, f32x8 c) {
  return __builtin_amdgcn_wmma_f32_16x16x32_bf16(false, a, false, b,
                                                 (short)0, c, false, false);
}

// bf16 A/B fragment from a contiguous 32-element K-slice.
// Lane-half 0 holds K={0..7,16..23}; half 1 holds K={8..15,24..31}.
__device__ __forceinline__ bf16x16 frag_bf16(const bf16_t* p, int half) {
  const int o = half * 8;
  bf16x8 lo = *(const bf16x8*)(p + o);
  bf16x8 hi = *(const bf16x8*)(p + o + 16);
  bf16x16 f;
#pragma unroll
  for (int i = 0; i < 8; ++i) { f[i] = lo[i]; f[8 + i] = hi[i]; }
  return f;
}

// Wave-level GEMM tile: C16x16 += A[row,:K] * W[col,:K]^T (both bf16).
__device__ __forceinline__ f32x8 wave_tile_gemm(const bf16_t* __restrict__ arow,
                                               const bf16_t* __restrict__ brow,
                                               int K, int half) {
  f32x8 c = {};
  for (int ks = 0; ks < K; ks += 32) {
    c = wmma_bf16(frag_bf16(arow + ks, half), frag_bf16(brow + ks, half), c);
  }
  return c;
}

// ---------------------------------------------------------------------------
// Kernel 0: one-shot fp32 -> bf16 conversion (query / W_qkv / W_out).
// ---------------------------------------------------------------------------
__global__ __launch_bounds__(256) void cvt_f32_bf16_kernel(
    const float* __restrict__ in, bf16_t* __restrict__ out) {
  const int i = (blockIdx.x * 256 + threadIdx.x) * 8;
  f32x4 a = *(const f32x4*)(in + i);
  f32x4 b = *(const f32x4*)(in + i + 4);
  bf16x8 o;
#pragma unroll
  for (int j = 0; j < 4; ++j) { o[j] = (bf16_t)a[j]; o[4 + j] = (bf16_t)b[j]; }
  *(bf16x8*)(out + i) = o;
}

// ---------------------------------------------------------------------------
// Kernel 1: QKV projection.  Xq[4096,1024]bf16 @ Wqkv[3072,1024]bf16^T + b.
// Row m = t*B + b.  Feature f: h=f/192, c=f%192 -> q/k/vT scatter (bf16).
// ---------------------------------------------------------------------------
__global__ __launch_bounds__(256) void qkv_gemm_kernel(
    const bf16_t* __restrict__ X, const bf16_t* __restrict__ W,
    const float* __restrict__ bias,
    bf16_t* __restrict__ qb, bf16_t* __restrict__ kb, bf16_t* __restrict__ vtb) {
  const int lane = threadIdx.x & 31;
  const int wave = threadIdx.x >> 5;
  const int half = lane >> 4;
  const int lcol = lane & 15;
  const int m0 = blockIdx.x * 32 + (wave >> 2) * 16;
  const int n0 = blockIdx.y * 64 + (wave & 3) * 16;

  const bf16_t* arow = X + (size_t)(m0 + lcol) * EMB;
  const bf16_t* brow = W + (size_t)(n0 + lcol) * EMB;
  f32x8 c = wave_tile_gemm(arow, brow, EMB, half);

  const int col = n0 + lcol;
  const float bv = bias[col];
  const int h = col / 192;
  const int cc = col % 192;
#pragma unroll
  for (int r = 0; r < 8; ++r) {
    const int mrow = m0 + half * 8 + r;       // C layout: VGPR r -> row r+8*half
    const bf16_t val = (bf16_t)(c[r] + bv);
    const int t = mrow >> 1;                  // B == 2
    const int b = mrow & 1;
    const int g = b * NHEAD + h;
    if (cc < 64)        qb[((g * T_LEN) + t) * HD + cc] = val;
    else if (cc < 128)  kb[((g * T_LEN) + t) * HD + (cc - 64)] = val;
    else                vtb[((g * HD) + (cc - 128)) * T_LEN + t] = val;
  }
}

// ---------------------------------------------------------------------------
// Kernel 2: fused flash-attention.  Block = 8 waves = 128 query rows of one
// head.  Shared K (32x64) / Vt (64x32) bf16 tiles are double-buffered in LDS:
// the copy of tile i+1 (TDM or async-to-LDS) is issued before computing tile
// i, so the DMA hides under S=QK^T + softmax + P@V.  One wait+barrier/step.
// ---------------------------------------------------------------------------
__global__ __launch_bounds__(256) void attn_kernel(
    const bf16_t* __restrict__ qb, const bf16_t* __restrict__ kb,
    const bf16_t* __restrict__ vtb, bf16_t* __restrict__ xbuf) {
  __shared__ bf16_t kt_lds[2][32 * HD];   // [buf][key][d]   2 x 4 KB
  __shared__ bf16_t vt_lds[2][HD * 32];   // [buf][d][key]   2 x 4 KB
  __shared__ bf16_t pl[8][16 * 32];       // per-wave P transpose tile, 8 KB

  const int tid  = threadIdx.x;
  const int lane = tid & 31;
  const int wave = tid >> 5;
  const int half = lane >> 4;
  const int lcol = lane & 15;
  const int g = blockIdx.y;
  const int q0 = blockIdx.x * 128 + wave * 16;

  const bf16_t* kg0 = kb + (size_t)g * T_LEN * HD;          // K rows for head g
  const bf16_t* vg0 = vtb + (size_t)g * HD * T_LEN;         // Vt rows for head g
#if !HAVE_TDM
  // Per-thread 16B chunk assignment for the cooperative async fallback.
  const int krow = tid >> 3, kcol = (tid & 7) * 8;          // K : 32 rows x 128B
  const int vrow = tid >> 2, vcol = (tid & 3) * 8;          // Vt: 64 rows x 64B
#endif

  // Stage K/Vt tiles for key block kt into LDS buffer b.
  auto stage = [&](int kt, int b) {
#if HAVE_TDM
    if (wave == 0) {
      tdm_load_2d(lds_offset(&kt_lds[b][0]),
                  (unsigned long long)(kg0 + (size_t)kt * HD),
                  /*rowBytes=*/HD * 2, /*rows=*/32, /*stride=*/HD * 2);
      tdm_load_2d(lds_offset(&vt_lds[b][0]),
                  (unsigned long long)(vg0 + kt),
                  /*rowBytes=*/32 * 2, /*rows=*/HD, /*stride=*/T_LEN * 2);
    }
#else
    copy16_to_lds(kg0 + (size_t)(kt + krow) * HD + kcol, &kt_lds[b][krow * HD + kcol]);
    copy16_to_lds(vg0 + (size_t)vrow * T_LEN + kt + vcol, &vt_lds[b][vrow * 32 + vcol]);
#endif
  };
  auto stage_wait = [&]() {
#if HAVE_TDM
    if (wave == 0) __builtin_amdgcn_s_wait_tensorcnt(0);
#else
    wait_async_copies();
#endif
  };

  // Q fragments (16 rows x 64 dims), fixed across the key loop.
  const bf16_t* qrow = qb + ((size_t)g * T_LEN + q0 + lcol) * HD;
  const bf16x16 aq0 = frag_bf16(qrow, half);
  const bf16x16 aq1 = frag_bf16(qrow + 32, half);

  f32x8 o0 = {}, o1 = {}, o2 = {}, o3 = {};
  float mrow[8], lrow[8];
#pragma unroll
  for (int r = 0; r < 8; ++r) { mrow[r] = NEG_INF; lrow[r] = 0.f; }

  // Prologue: fill buffer 0.
  stage(0, 0);
  stage_wait();
  __syncthreads();

  for (int kt = 0; kt < T_LEN; kt += 32) {
    const int cur = (kt >> 5) & 1;
    if (kt + 32 < T_LEN) stage(kt + 32, cur ^ 1);   // overlap DMA with compute

    // --- S = Q @ K^T (16 x 32 scores) ---
    const bf16_t* kr0 = &kt_lds[cur][lcol * HD];
    const bf16_t* kr1 = &kt_lds[cur][(16 + lcol) * HD];
    f32x8 s0 = {}, s1 = {};
    s0 = wmma_bf16(aq0, frag_bf16(kr0, half), s0);
    s0 = wmma_bf16(aq1, frag_bf16(kr0 + 32, half), s0);
    s1 = wmma_bf16(aq0, frag_bf16(kr1, half), s1);
    s1 = wmma_bf16(aq1, frag_bf16(kr1 + 32, half), s1);

    // scale + reference quirk: exact-zero scores -> -inf
#pragma unroll
    for (int r = 0; r < 8; ++r) {
      s0[r] = (s0[r] == 0.f) ? NEG_INF : s0[r] * 0.125f;
      s1[r] = (s1[r] == 0.f) ? NEG_INF : s1[r] * 0.125f;
    }

    // --- online softmax (row stats per C-VGPR, reduce across 16 lanes) ---
#pragma unroll
    for (int r = 0; r < 8; ++r) {
      float v = fmaxf(s0[r], s1[r]);
      v = fmaxf(v, __shfl_xor(v, 1));
      v = fmaxf(v, __shfl_xor(v, 2));
      v = fmaxf(v, __shfl_xor(v, 4));
      v = fmaxf(v, __shfl_xor(v, 8));
      const float mn = fmaxf(mrow[r], v);
      const float al = __expf(mrow[r] - mn);
      const float e0 = __expf(s0[r] - mn);
      const float e1 = __expf(s1[r] - mn);
      s0[r] = e0; s1[r] = e1;
      float rs = e0 + e1;
      rs += __shfl_xor(rs, 1);
      rs += __shfl_xor(rs, 2);
      rs += __shfl_xor(rs, 4);
      rs += __shfl_xor(rs, 8);
      lrow[r] = lrow[r] * al + rs;
      mrow[r] = mn;
      o0[r] *= al; o1[r] *= al; o2[r] *= al; o3[r] *= al;
    }

    // --- P: C-layout -> A-layout via per-wave LDS tile ---
    bf16_t* pw = &pl[wave][0];
#pragma unroll
    for (int r = 0; r < 8; ++r) {
      const int row = r + 8 * half;
      pw[row * 32 + lcol]      = (bf16_t)s0[r];
      pw[row * 32 + 16 + lcol] = (bf16_t)s1[r];
    }
    asm volatile("s_wait_dscnt 0" ::: "memory");
    const bf16_t* pr = &pl[wave][lcol * 32];
    bf16x8 plo = *(const bf16x8*)(pr + half * 8);
    bf16x8 phi = *(const bf16x8*)(pr + half * 8 + 16);
    bf16x16 ap;
#pragma unroll
    for (int i = 0; i < 8; ++i) { ap[i] = plo[i]; ap[8 + i] = phi[i]; }

    // --- O += P @ V  (Vt tile: B columns are LDS rows) ---
    const bf16_t* vr = &vt_lds[cur][lcol * 32];
    o0 = wmma_bf16(ap, frag_bf16(vr, half), o0);
    o1 = wmma_bf16(ap, frag_bf16(vr + 16 * 32, half), o1);
    o2 = wmma_bf16(ap, frag_bf16(vr + 32 * 32, half), o2);
    o3 = wmma_bf16(ap, frag_bf16(vr + 48 * 32, half), o3);

    // Next tile's DMA must have landed; readers of `cur` are done after this.
    stage_wait();
    __syncthreads();
  }

  // --- normalize + scatter (bf16) into X with the reference reshape quirk:
  //     b = t//1024, t2 = 2*(t%1024) + (g//16), e = (g%16)*64 + d
  const int b0 = g >> 4;
  const int h = g & 15;
#pragma unroll
  for (int r = 0; r < 8; ++r) {
    const int t = q0 + r + 8 * half;
    const float inv = 1.f / lrow[r];
    const int xrow = (t >> 10) * (T_LEN * EMB) + (2 * (t & 1023) + b0) * EMB + h * HD;
    xbuf[xrow + lcol]      = (bf16_t)(o0[r] * inv);
    xbuf[xrow + 16 + lcol] = (bf16_t)(o1[r] * inv);
    xbuf[xrow + 32 + lcol] = (bf16_t)(o2[r] * inv);
    xbuf[xrow + 48 + lcol] = (bf16_t)(o3[r] * inv);
  }
}

// ---------------------------------------------------------------------------
// Kernel 3: output projection.  Out[4096,1024]f32 = Xbf16 @ Wout_bf16^T + b.
// ---------------------------------------------------------------------------
__global__ __launch_bounds__(256) void out_gemm_kernel(
    const bf16_t* __restrict__ X, const bf16_t* __restrict__ W,
    const float* __restrict__ bias, float* __restrict__ out) {
  const int lane = threadIdx.x & 31;
  const int wave = threadIdx.x >> 5;
  const int half = lane >> 4;
  const int lcol = lane & 15;
  const int m0 = blockIdx.x * 32 + (wave >> 2) * 16;
  const int n0 = blockIdx.y * 64 + (wave & 3) * 16;

  const bf16_t* arow = X + (size_t)(m0 + lcol) * EMB;
  const bf16_t* brow = W + (size_t)(n0 + lcol) * EMB;
  f32x8 c = wave_tile_gemm(arow, brow, EMB, half);

  const int col = n0 + lcol;
  const float bv = bias[col];
#pragma unroll
  for (int r = 0; r < 8; ++r) {
    const int mrow = m0 + half * 8 + r;
    out[(size_t)mrow * EMB + col] = c[r] + bv;
  }
}

// ---------------------------------------------------------------------------
extern "C" void kernel_launch(void* const* d_in, const int* in_sizes, int n_in,
                              void* d_out, int out_size, void* d_ws, size_t ws_size,
                              hipStream_t stream) {
  const float* query = (const float*)d_in[0];
  const float* W_qkv = (const float*)d_in[5];
  const float* b_qkv = (const float*)d_in[6];
  const float* W_out = (const float*)d_in[7];
  const float* b_out = (const float*)d_in[8];
  float* out = (float*)d_out;

  // bf16 workspace carve (all sizes 16B-aligned):
  bf16_t* Xq   = (bf16_t*)d_ws;                         //  4096*1024  (8 MB)
  bf16_t* Wq   = Xq  + (size_t)4096 * EMB;              //  3072*1024  (6 MB)
  bf16_t* Wo   = Wq  + (size_t)3072 * EMB;              //  1024*1024  (2 MB)
  bf16_t* qbuf = Wo  + (size_t)1024 * EMB;              //  BH*T*HD    (8 MB)
  bf16_t* kbuf = qbuf + (size_t)BH * T_LEN * HD;
  bf16_t* vtb  = kbuf + (size_t)BH * T_LEN * HD;
  bf16_t* xbuf = vtb  + (size_t)BH * T_LEN * HD;        //  B*T*E      (8 MB)

  const dim3 blk(256);
  // fp32 -> bf16 producers (each thread converts 8 elements)
  cvt_f32_bf16_kernel<<<dim3((4096 * EMB) / 2048), blk, 0, stream>>>(query, Xq);
  cvt_f32_bf16_kernel<<<dim3((3072 * EMB) / 2048), blk, 0, stream>>>(W_qkv, Wq);
  cvt_f32_bf16_kernel<<<dim3((1024 * EMB) / 2048), blk, 0, stream>>>(W_out, Wo);

  qkv_gemm_kernel<<<dim3(128, 48), blk, 0, stream>>>(Xq, Wq, b_qkv,
                                                     qbuf, kbuf, vtb);
  attn_kernel<<<dim3(16, 32), blk, 0, stream>>>(qbuf, kbuf, vtb, xbuf);
  out_gemm_kernel<<<dim3(128, 16), blk, 0, stream>>>(xbuf, Wo, b_out, out);
}
